// GatedSparseAttention_72069551227501
// MI455X (gfx1250) — compile-verified, matrix operand order checked by reading, not run on hardware
//
#include <hip/hip_runtime.h>
#include <hip/hip_bf16.h>

typedef __bf16 bf16;
typedef __attribute__((ext_vector_type(8)))  float v8f;
typedef __attribute__((ext_vector_type(8)))  bf16  v8bf;
typedef __attribute__((ext_vector_type(16))) bf16  v16bf;

#define B_   2
#define S_   2049
#define E_   512
#define H_   8
#define D_   64
#define WIN_ 1024
#define SP_  2080   // padded seq: multiple of 32, >= 129*16 and >= last key-block end
#define QT_  129    // query tiles of 16: 129*16 = 2064 >= 2049
#define MP_  4160   // B_*S_ = 4098 padded to multiple of 64 (GEMM row tiles)

// ---------------------------------------------------------------------------
// Fragment loader: 16-bit A/B fragment for v_wmma_*_16x16x32_bf16.
// lane = (m or n) = laneid%16, half = laneid/16.
// Elements 0..7  -> K = 8*half + 0..7   (contiguous 16B)
// Elements 8..15 -> K = 16 + 8*half + 0..7 (contiguous 16B)
// ---------------------------------------------------------------------------
__device__ __forceinline__ v16bf load_frag(const bf16* rowptr, int half) {
  const v8bf lo = *(const v8bf*)(rowptr + 8 * half);
  const v8bf hi = *(const v8bf*)(rowptr + 16 + 8 * half);
  v16bf r;
#pragma unroll
  for (int i = 0; i < 8; ++i) { r[i] = lo[i]; r[8 + i] = hi[i]; }
  return r;
}

__device__ __forceinline__ v8f wmma_bf16(v16bf a, v16bf b, v8f c) {
  return __builtin_amdgcn_wmma_f32_16x16x32_bf16(false, a, false, b, (short)0, c,
                                                 false, false);
}

// CDNA5 async memory->LDS copy, 16B per lane. lds_off = LDS byte offset
// (low 32 bits of a generic pointer to LDS), g = per-lane global address.
// Tracked by ASYNCcnt; completes in order within a wave.
__device__ __forceinline__ void async_copy16(unsigned lds_off, const void* g) {
  asm volatile("global_load_async_to_lds_b128 %0, %1, off"
               :: "v"(lds_off), "v"(g) : "memory");
}

// ---------------------------------------------------------------------------
// f32 -> bf16 conversion (activations)
// ---------------------------------------------------------------------------
__global__ void f32_to_bf16_kernel(const float* __restrict__ in,
                                   bf16* __restrict__ out, int n) {
  int i = blockIdx.x * blockDim.x + threadIdx.x;
  if (i < n) out[i] = (bf16)in[i];
}

// f32 [K,N] -> bf16 transposed [N,K] (weights; lets GEMM B-tiles stage as
// contiguous rows through the async-to-LDS path)
__global__ void transpose_to_bf16_kernel(const float* __restrict__ in,
                                         bf16* __restrict__ out, int K, int N) {
  int i = blockIdx.x * blockDim.x + threadIdx.x;
  if (i < K * N) {
    int k = i / N, n = i - k * N;
    out[(size_t)n * K + k] = (bf16)in[i];
  }
}

// ---------------------------------------------------------------------------
// Tiled WMMA GEMM:  C[M,N] = A[M,K](bf16) @ Wt[N,K]^T(bf16) + bias
// Block = 128 threads (4 waves), 64x64 tile; each wave owns a 32x32 quadrant.
// A and B tiles are double-buffered in LDS via GLOBAL_LOAD_ASYNC_TO_LDS_B128
// (4 async instructions per wave per tile-pair; s_wait_asynccnt 4 overlaps the
// next tile's DMA with this tile's WMMAs). A must have >= ceil64(M) rows.
// mode 0: scatter bf16 to [b,h,s,d]  (q/k, padded seq SP_)
// mode 1: scatter bf16 to [b,h,d,s]  (v transposed)
// mode 2: f32 row-major [M,N] (final output)
// ---------------------------------------------------------------------------
__global__ __launch_bounds__(128) void gemm_wmma_kernel(
    const bf16* __restrict__ A, int M, const bf16* __restrict__ Wt,
    const float* __restrict__ bias, int N, int K, int mode,
    void* __restrict__ outp) {
  __shared__ __align__(16) bf16 As[2][64 * 32];
  __shared__ __align__(16) bf16 Bs[2][64 * 32];  // [n][k]

  const int tiles_n = N >> 6;
  const int m0 = (blockIdx.x / tiles_n) << 6;
  const int n0 = (blockIdx.x % tiles_n) << 6;
  const int t = threadIdx.x;
  const int warp = t >> 5, lane = t & 31;
  const int m16 = lane & 15, half = lane >> 4;
  const int wm = (warp >> 1) << 5;
  const int wn = (warp & 1) << 5;

  // per-lane staging assignment: row = t/2, 16-element column half
  const int sr = t >> 1;
  const int sc = (t & 1) << 4;
  const bf16* gA = A + (size_t)(m0 + sr) * K + sc;
  const bf16* gB = Wt + (size_t)(n0 + sr) * K + sc;
  const unsigned laA0 = (unsigned)(uintptr_t)(&As[0][sr * 32 + sc]);
  const unsigned laB0 = (unsigned)(uintptr_t)(&Bs[0][sr * 32 + sc]);
  const unsigned bufstep = (unsigned)(64 * 32 * sizeof(bf16));

  v8f acc[2][2];
#pragma unroll
  for (int mi = 0; mi < 2; ++mi)
#pragma unroll
    for (int ni = 0; ni < 2; ++ni)
#pragma unroll
      for (int r = 0; r < 8; ++r) acc[mi][ni][r] = 0.f;

  // prologue: stage tile 0 into buffer 0
  async_copy16(laA0, gA);
  async_copy16(laA0 + 16, gA + 8);
  async_copy16(laB0, gB);
  async_copy16(laB0 + 16, gB + 8);

  int buf = 0;
  for (int kk = 0; kk < K; kk += 32, buf ^= 1) {
    const bool more = (kk + 32) < K;
    if (more) {  // prefetch next tile into the other buffer
      unsigned la = laA0 + (buf ^ 1) * bufstep;
      unsigned lb = laB0 + (buf ^ 1) * bufstep;
      const bf16* ga = gA + kk + 32;
      const bf16* gb = gB + kk + 32;
      async_copy16(la, ga);
      async_copy16(la + 16, ga + 8);
      async_copy16(lb, gb);
      async_copy16(lb + 16, gb + 8);
      asm volatile("s_wait_asynccnt 4" ::: "memory");  // current tile landed
    } else {
      asm volatile("s_wait_asynccnt 0" ::: "memory");
    }
    __syncthreads();

    v16bf af[2], bfrag[2];
    af[0] = load_frag(&As[buf][(wm + m16) * 32], half);
    af[1] = load_frag(&As[buf][(wm + 16 + m16) * 32], half);
    bfrag[0] = load_frag(&Bs[buf][(wn + m16) * 32], half);
    bfrag[1] = load_frag(&Bs[buf][(wn + 16 + m16) * 32], half);
#pragma unroll
    for (int mi = 0; mi < 2; ++mi)
#pragma unroll
      for (int ni = 0; ni < 2; ++ni)
        acc[mi][ni] = wmma_bf16(af[mi], bfrag[ni], acc[mi][ni]);
    __syncthreads();  // protect buffer being re-filled next iteration
  }

#pragma unroll
  for (int mi = 0; mi < 2; ++mi)
#pragma unroll
    for (int ni = 0; ni < 2; ++ni) {
      int col = n0 + wn + ni * 16 + m16;
      float bv = bias[col];
#pragma unroll
      for (int r = 0; r < 8; ++r) {
        int row = m0 + wm + mi * 16 + r + 8 * half;
        if (row < M) {
          float val = acc[mi][ni][r] + bv;
          if (mode == 2) {
            ((float*)outp)[(size_t)row * N + col] = val;
          } else {
            int bb = row / S_, s = row - bb * S_;
            int hh = col >> 6, dd = col & 63;
            bf16* o = (bf16*)outp;
            if (mode == 0)
              o[(((size_t)(bb * H_ + hh)) * SP_ + s) * D_ + dd] = (bf16)val;
            else
              o[(((size_t)(bb * H_ + hh)) * D_ + dd) * SP_ + s] = (bf16)val;
          }
        }
      }
    }
}

// ---------------------------------------------------------------------------
// Gate projections: gq[b,s] = q_flat . w_gate[:E], gk[b,s] = k_flat . w_gate[E:]
// ---------------------------------------------------------------------------
__global__ void gate_kernel(const bf16* __restrict__ qb,
                            const bf16* __restrict__ kb,
                            const float* __restrict__ wg,
                            float* __restrict__ gq, float* __restrict__ gk) {
  int t = blockIdx.x * blockDim.x + threadIdx.x;
  if (t >= B_ * S_) return;
  int b = t / S_, s = t - b * S_;
  float aq = 0.f, ak = 0.f;
  for (int h = 0; h < H_; ++h) {
    const bf16* qrow = qb + (((size_t)(b * H_ + h)) * SP_ + s) * D_;
    const bf16* krow = kb + (((size_t)(b * H_ + h)) * SP_ + s) * D_;
    const float* wq = wg + h * D_;
    const float* wk = wg + E_ + h * D_;
    for (int d = 0; d < D_; ++d) {
      aq += (float)qrow[d] * wq[d];
      ak += (float)krow[d] * wk[d];
    }
  }
  gq[b * SP_ + s] = aq;
  gk[b * SP_ + s] = ak;
}

// ---------------------------------------------------------------------------
// Flash-style windowed attention. One wave per 16-query tile per (b,h).
// Per 32-key block: scores via 4 bf16 WMMAs, bias+gate+mask epilogue,
// online softmax (shfl_xor row reductions aligned with the C/D layout),
// P staged bf16 in per-wave LDS, PV via 4 bf16 WMMAs from V^T.
// ---------------------------------------------------------------------------
__global__ __launch_bounds__(128) void attn_kernel(
    const bf16* __restrict__ qb, const bf16* __restrict__ kb,
    const bf16* __restrict__ vtb, const float* __restrict__ gq,
    const float* __restrict__ gk, const float* __restrict__ bgp,
    bf16* __restrict__ attnout) {
  __shared__ __align__(16) bf16 smem[4][16 * 64];
  const int warp = threadIdx.x >> 5, lane = threadIdx.x & 31;
  const int wid = blockIdx.x * 4 + warp;
  if (wid >= 16 * QT_) return;
  const int qt = wid % QT_, bh = wid / QT_;
  const int b = bh >> 3, h = bh & 7;
  const int q0 = qt << 4;
  const int m16 = lane & 15, half = lane >> 4;
  bf16* Pl = &smem[warp][0];

  const bf16* Qb = qb + (size_t)bh * SP_ * D_;
  const bf16* Kb = kb + (size_t)bh * SP_ * D_;
  const bf16* Vb = vtb + (size_t)bh * D_ * SP_;

  v16bf qf[2];
  qf[0] = load_frag(Qb + (size_t)(q0 + m16) * D_, half);
  qf[1] = load_frag(Qb + (size_t)(q0 + m16) * D_ + 32, half);

  float gqv[8];
  {
    const float* g = gq + b * SP_ + q0 + 8 * half;
#pragma unroll
    for (int r = 0; r < 8; ++r) gqv[r] = g[r];
  }
  const float bg = bgp[0];

  float mrun[8], lrun[8];
  v8f o[4];
#pragma unroll
  for (int r = 0; r < 8; ++r) { mrun[r] = -1.0e30f; lrun[r] = 0.f; }
#pragma unroll
  for (int dt = 0; dt < 4; ++dt)
#pragma unroll
    for (int r = 0; r < 8; ++r) o[dt][r] = 0.f;

  int kstart = q0 - WIN_;
  if (kstart < 0) kstart = 0;
  kstart &= ~31;
  int kend = q0 + 15 + WIN_;
  if (kend > S_ - 1) kend = S_ - 1;

  const float scale = 0.125f;                  // D^-0.5
  const float bcoef = 0.5f / (512.f * 512.f);  // 0.5/sigma^2, sigma = WIN/2

  for (int k0 = kstart; k0 <= kend; k0 += 32) {
    v8f sc[2];
#pragma unroll
    for (int jt = 0; jt < 2; ++jt) {
      const bf16* krow = Kb + (size_t)(k0 + jt * 16 + m16) * D_;
      v16bf kf0 = load_frag(krow, half);
      v16bf kf1 = load_frag(krow + 32, half);
      v8f c;
#pragma unroll
      for (int r = 0; r < 8; ++r) c[r] = 0.f;
      c = wmma_bf16(qf[0], kf0, c);
      c = wmma_bf16(qf[1], kf1, c);
      sc[jt] = c;
    }

    float gkj[2];
    gkj[0] = gk[b * SP_ + k0 + m16];
    gkj[1] = gk[b * SP_ + k0 + 16 + m16];

    float sv[2][8], okf[2][8], tmax[8];
#pragma unroll
    for (int r = 0; r < 8; ++r) tmax[r] = -1.0e30f;
#pragma unroll
    for (int jt = 0; jt < 2; ++jt) {
      int j = k0 + jt * 16 + m16;
#pragma unroll
      for (int r = 0; r < 8; ++r) {
        int i = q0 + r + 8 * half;
        float dd = (float)(i - j);
        float val = sc[jt][r] * scale - dd * dd * bcoef;
        float g = 1.f / (1.f + __expf(-(gqv[r] + gkj[jt] + bg)));
        val *= g;
        bool ok = (dd <= 1024.f) && (dd >= -1024.f) && (j < S_);
        sv[jt][r] = ok ? val : -1.0e30f;
        okf[jt][r] = ok ? 1.f : 0.f;
        tmax[r] = fmaxf(tmax[r], sv[jt][r]);
      }
    }
    // row-max across the 16 lanes holding each row (xor<16 stays in half)
#pragma unroll
    for (int off = 1; off < 16; off <<= 1)
#pragma unroll
      for (int r = 0; r < 8; ++r)
        tmax[r] = fmaxf(tmax[r], __shfl_xor(tmax[r], off, 32));

    float corr[8], rs[8];
#pragma unroll
    for (int r = 0; r < 8; ++r) {
      float nm = fmaxf(mrun[r], tmax[r]);
      corr[r] = __expf(mrun[r] - nm);
      mrun[r] = nm;
      rs[r] = 0.f;
    }
#pragma unroll
    for (int jt = 0; jt < 2; ++jt)
#pragma unroll
      for (int r = 0; r < 8; ++r) {
        float pv = okf[jt][r] * __expf(sv[jt][r] - mrun[r]);
        rs[r] += pv;
        Pl[(r + 8 * half) * 32 + jt * 16 + m16] = (bf16)pv;
      }
#pragma unroll
    for (int off = 1; off < 16; off <<= 1)
#pragma unroll
      for (int r = 0; r < 8; ++r) rs[r] += __shfl_xor(rs[r], off, 32);
#pragma unroll
    for (int r = 0; r < 8; ++r) lrun[r] = lrun[r] * corr[r] + rs[r];
#pragma unroll
    for (int dt = 0; dt < 4; ++dt)
#pragma unroll
      for (int r = 0; r < 8; ++r) o[dt][r] *= corr[r];

    // wave-local LDS RAW fence (DS ops are in-order within a wave)
    __asm__ volatile("s_wait_dscnt 0" ::: "memory");
    v16bf pf = load_frag(Pl + m16 * 32, half);
#pragma unroll
    for (int dt = 0; dt < 4; ++dt) {
      const bf16* vrow = Vb + (size_t)(dt * 16 + m16) * SP_ + k0;
      v16bf vf = load_frag(vrow, half);
      o[dt] = wmma_bf16(pf, vf, o[dt]);
    }
  }

  // finalize: divide by row sums, stage bf16 tile in LDS, coalesced store
#pragma unroll
  for (int dt = 0; dt < 4; ++dt)
#pragma unroll
    for (int r = 0; r < 8; ++r) {
      float inv = (lrun[r] > 0.f) ? 1.f / lrun[r] : 0.f;
      Pl[(r + 8 * half) * 64 + dt * 16 + m16] = (bf16)(o[dt][r] * inv);
    }
  __asm__ volatile("s_wait_dscnt 0" ::: "memory");
  {
    int row = lane >> 1;
    int c0 = (lane & 1) << 5;
    int s = q0 + row;
    if (s < S_) {
      bf16* dst = attnout + ((size_t)(b * S_ + s)) * E_ + h * D_ + c0;
      const v8bf* src = (const v8bf*)(Pl + row * 64 + c0);
      v8bf* d4 = (v8bf*)dst;
#pragma unroll
      for (int i = 0; i < 4; ++i) d4[i] = src[i];
    }
  }
}

// ---------------------------------------------------------------------------
extern "C" void kernel_launch(void* const* d_in, const int* in_sizes, int n_in,
                              void* d_out, int out_size, void* d_ws,
                              size_t ws_size, hipStream_t stream) {
  (void)in_sizes; (void)n_in; (void)out_size; (void)ws_size;
  const float* x  = (const float*)d_in[0];
  const float* Wq = (const float*)d_in[1];
  const float* bq = (const float*)d_in[2];
  const float* Wk = (const float*)d_in[3];
  const float* bk = (const float*)d_in[4];
  const float* Wv = (const float*)d_in[5];
  const float* bv = (const float*)d_in[6];
  const float* Wo = (const float*)d_in[7];
  const float* bo = (const float*)d_in[8];
  const float* wg = (const float*)d_in[9];
  const float* bg = (const float*)d_in[10];

  char* ws = (char*)d_ws;
  size_t off = 0;
  auto alloc = [&](size_t bytes) -> void* {
    void* p = ws + off;
    off += (bytes + 255) & ~(size_t)255;
    return p;
  };

  const size_t headbuf = (size_t)B_ * H_ * SP_ * D_ * sizeof(bf16);
  bf16* qbuf  = (bf16*)alloc(headbuf);
  bf16* kbuf  = (bf16*)alloc(headbuf);
  bf16* vtbuf = (bf16*)alloc(headbuf);
  float* gqv  = (float*)alloc((size_t)B_ * SP_ * sizeof(float));
  float* gkv  = (float*)alloc((size_t)B_ * SP_ * sizeof(float));
  const size_t zbytes = off;  // zero-filled region (covers seq padding)
  // M padded to 64 so async tile loads never run past the buffer
  bf16* xb   = (bf16*)alloc((size_t)MP_ * E_ * sizeof(bf16));
  bf16* aout = (bf16*)alloc((size_t)MP_ * E_ * sizeof(bf16));
  bf16* Wqt = (bf16*)alloc((size_t)E_ * E_ * sizeof(bf16));
  bf16* Wkt = (bf16*)alloc((size_t)E_ * E_ * sizeof(bf16));
  bf16* Wvt = (bf16*)alloc((size_t)E_ * E_ * sizeof(bf16));
  bf16* Wot = (bf16*)alloc((size_t)E_ * E_ * sizeof(bf16));

  hipMemsetAsync(d_ws, 0, zbytes, stream);

  const int ncv = B_ * S_ * E_;
  f32_to_bf16_kernel<<<(ncv + 255) / 256, 256, 0, stream>>>(x, xb, ncv);
  const int nw = E_ * E_;
  transpose_to_bf16_kernel<<<(nw + 255) / 256, 256, 0, stream>>>(Wq, Wqt, E_, E_);
  transpose_to_bf16_kernel<<<(nw + 255) / 256, 256, 0, stream>>>(Wk, Wkt, E_, E_);
  transpose_to_bf16_kernel<<<(nw + 255) / 256, 256, 0, stream>>>(Wv, Wvt, E_, E_);
  transpose_to_bf16_kernel<<<(nw + 255) / 256, 256, 0, stream>>>(Wo, Wot, E_, E_);

  const int M = B_ * S_;
  dim3 ggrid((MP_ / 64) * (E_ / 64));
  gemm_wmma_kernel<<<ggrid, 128, 0, stream>>>(xb, M, Wqt, bq, E_, E_, 0, qbuf);
  gemm_wmma_kernel<<<ggrid, 128, 0, stream>>>(xb, M, Wkt, bk, E_, E_, 0, kbuf);
  gemm_wmma_kernel<<<ggrid, 128, 0, stream>>>(xb, M, Wvt, bv, E_, E_, 1, vtbuf);

  gate_kernel<<<(M + 255) / 256, 256, 0, stream>>>(qbuf, kbuf, wg, gqv, gkv);

  attn_kernel<<<(16 * QT_) / 4, 128, 0, stream>>>(qbuf, kbuf, vtbuf, gqv, gkv,
                                                  bg, aout);

  gemm_wmma_kernel<<<ggrid, 128, 0, stream>>>(aout, M, Wot, bo, E_, E_, 2,
                                              d_out);
}
